// TextProjectModel_48747878809821
// MI455X (gfx1250) — matrix operand order repaired and checked by reading, not compiled, not run on hardware
//
#include <hip/hip_runtime.h>
#include <hip/hip_bf16.h>

// ---------------- model constants ----------------
constexpr int kB  = 2;
constexpr int kS  = 4096;
constexpr int kD  = 768;
constexpr int kH  = 12;
constexpr int kDH = 64;
constexpr int kL  = 12;
constexpr int kFF = 3072;
constexpr int kW  = 256;
constexpr int kNCH = kS / kW;        // 16 chunks
constexpr int kSP  = kS + 2 * kW;    // padded key length 4608
constexpr int kKW  = 3 * kW;         // 768 keys per window

typedef __bf16 bf16;
typedef __bf16 v16bf __attribute__((ext_vector_type(16)));
typedef float  v8f   __attribute__((ext_vector_type(8)));
typedef unsigned int u32x4 __attribute__((ext_vector_type(4)));
typedef int i32x4 __attribute__((ext_vector_type(4)));
typedef int i32x8 __attribute__((ext_vector_type(8)));

#define WMMA_BF16(a, b, c) \
  __builtin_amdgcn_wmma_f32_16x16x32_bf16(false, (a), false, (b), (short)0, (c), false, false)

#if __has_builtin(__builtin_amdgcn_tensor_load_to_lds) && \
    __has_builtin(__builtin_amdgcn_s_wait_tensorcnt)
#define USE_TDM 1
#else
#define USE_TDM 0
#endif

__device__ __forceinline__ bf16 tobf(float x) {
  // round-to-nearest-even fp32 -> bf16
  union { float f; unsigned u; } a; a.f = x;
  unsigned r = a.u + 0x7fffu + ((a.u >> 16) & 1u);
  union { unsigned short s; bf16 b; } o; o.s = (unsigned short)(r >> 16);
  return o.b;
}

union FragU { v16bf v; uint4 q[2]; };

// A-matrix 16x32 bf16 fragment (staggered K per ISA table):
// lanes 0-15: M=lane, K = {0..7, 16..23}; lanes 16-31: M=lane-16, K = {8..15, 24..31}
__device__ __forceinline__ v16bf load_fragA(const bf16* base, int stride, int rc, int k0) {
  int lane = threadIdx.x & 31;
  const bf16* p = base + (size_t)(rc + (lane & 15)) * stride + k0 + ((lane >> 4) << 3);
  FragU u;
  u.q[0] = *(const uint4*)(p);
  u.q[1] = *(const uint4*)(p + 16);
  return u.v;
}

// B-matrix 32x16 bf16 fragment (staggered K, matching A):
// lanes 0-15: N=lane, K = {0..7, 16..23}; lanes 16-31: N=lane-16, K = {8..15, 24..31}.  base[n*stride + k]
__device__ __forceinline__ v16bf load_fragB(const bf16* base, int stride, int rc, int k0) {
  int lane = threadIdx.x & 31;
  const bf16* p = base + (size_t)(rc + (lane & 15)) * stride + k0 + ((lane >> 4) << 3);
  FragU u;
  u.q[0] = *(const uint4*)(p);
  u.q[1] = *(const uint4*)(p + 16);
  return u.v;
}

#if USE_TDM
// Issue a TDM 2D tile load: global bf16 tile (rows x kcols, row stride = stride_elems)
// -> contiguous LDS [rows][kcols]. D# bitfields per CDNA5 ISA ch.8.
__device__ __forceinline__ void tdm_load_tile(unsigned lds_off, const bf16* gptr,
                                              int rows, int kcols, int stride_elems) {
  unsigned long long ga = (unsigned long long)(uintptr_t)gptr;
  unsigned td0 = (unsigned)stride_elems;     // tensor_dim0 (row length, elems)
  unsigned td1 = (unsigned)rows;             // tensor_dim1 (enough rows for OOB)
  unsigned long long st0 = (unsigned long long)stride_elems;
  u32x4 g0;
  g0[0] = 1u;                                             // count=1, user mode
  g0[1] = lds_off;                                        // LDS byte address
  g0[2] = (unsigned)(ga & 0xffffffffull);                 // global_addr[31:0]
  g0[3] = (unsigned)((ga >> 32) & 0x01ffffffull) | (2u << 30);  // addr[56:32] | type=2
  i32x8 g1;
  g1[0] = (int)(1u << 16);                                // data_size=1 (2 bytes)
  g1[1] = (int)((td0 & 0xffffu) << 16);                   // tensor_dim0[15:0] @ bits 63:48
  g1[2] = (int)(((td0 >> 16) & 0xffffu) | ((td1 & 0xffffu) << 16));
  g1[3] = (int)(((td1 >> 16) & 0xffffu) | ((unsigned)kcols << 16)); // tile_dim0
  g1[4] = (int)((unsigned)rows & 0xffffu);                // tile_dim1 (tile_dim2=0)
  g1[5] = (int)(st0 & 0xffffffffull);                     // tensor_dim0_stride[31:0]
  g1[6] = (int)((st0 >> 32) & 0xffffull);                 // stride[47:32], dim1_stride=0
  g1[7] = 0;
  i32x4 z4 = {0, 0, 0, 0};
#if defined(__clang_major__) && __clang_major__ >= 23
  i32x8 z8 = {0, 0, 0, 0, 0, 0, 0, 0};
  __builtin_amdgcn_tensor_load_to_lds(g0, g1, z4, z4, z8, 0);
#else
  __builtin_amdgcn_tensor_load_to_lds(g0, g1, z4, z4, 0);
#endif
}
#endif

// ---------------- GEMM: C[M,N] = act(A[M,K] @ Bt[N,K]^T + bias) ----------------
// bf16 operands (A row-major [M][K], B pre-transposed [N][K]), fp32 accumulate.
// Block tile 128x64, K-step 64, 8 waves, wave tile 32x32. Double-buffered TDM staging.
constexpr int GBM = 128, GBN = 64, GBK = 64;

__global__ __launch_bounds__(256) void gemm_bf16_kernel(
    const bf16* __restrict__ A, const bf16* __restrict__ Bt,
    const float* __restrict__ bias, float* __restrict__ C, bf16* __restrict__ Cb,
    int M, int N, int K, int act)
{
  __shared__ __align__(16) bf16 sA[2][GBM][GBK];    // 32 KB
  __shared__ __align__(16) bf16 sB[2][GBN][GBK];    // 16 KB
  const int bm = blockIdx.y * GBM, bn = blockIdx.x * GBN;
  const int tid = threadIdx.x;
  const int wid = tid >> 5;
  const int wm = (wid >> 1) * 32, wn = (wid & 1) * 32;
  const int nk = K / GBK;
  v8f acc[2][2] = {};

#if USE_TDM
  if (tid < 32) {
    tdm_load_tile((unsigned)(uintptr_t)&sA[0][0][0], A + (size_t)bm * K, GBM, GBK, K);
    tdm_load_tile((unsigned)(uintptr_t)&sB[0][0][0], Bt + (size_t)bn * K, GBN, GBK, K);
  }
  for (int it = 0; it < nk; ++it) {
    const int cur = it & 1;
    if (tid < 32) {
      if (it + 1 < nk) {
        const int kt = (it + 1) * GBK;
        tdm_load_tile((unsigned)(uintptr_t)&sA[cur ^ 1][0][0], A + (size_t)bm * K + kt, GBM, GBK, K);
        tdm_load_tile((unsigned)(uintptr_t)&sB[cur ^ 1][0][0], Bt + (size_t)bn * K + kt, GBN, GBK, K);
        __builtin_amdgcn_s_wait_tensorcnt((short)2);   // current tile's 2 ops done
      } else {
        __builtin_amdgcn_s_wait_tensorcnt((short)0);
      }
    }
    __syncthreads();                                   // tile ready for all waves
    for (int kk = 0; kk < GBK; kk += 32) {
      v16bf a0 = load_fragA(&sA[cur][0][0], GBK, wm, kk);
      v16bf a1 = load_fragA(&sA[cur][0][0], GBK, wm + 16, kk);
      v16bf b0 = load_fragB(&sB[cur][0][0], GBK, wn, kk);
      v16bf b1 = load_fragB(&sB[cur][0][0], GBK, wn + 16, kk);
      acc[0][0] = WMMA_BF16(a0, b0, acc[0][0]);
      acc[0][1] = WMMA_BF16(a0, b1, acc[0][1]);
      acc[1][0] = WMMA_BF16(a1, b0, acc[1][0]);
      acc[1][1] = WMMA_BF16(a1, b1, acc[1][1]);
    }
    __syncthreads();                                   // safe to overwrite this buffer
  }
#else
  for (int it = 0; it < nk; ++it) {
    const int cur = it & 1;
    const int kt = it * GBK;
    // manual staging: contiguous 16B copies (bf16 -> bf16)
    for (int i = tid; i < GBM * GBK / 8; i += 256) {
      int row = i >> 3, cc = (i & 7) << 3;
      *(uint4*)&sA[cur][row][cc] = *(const uint4*)(A + (size_t)(bm + row) * K + kt + cc);
    }
    for (int i = tid; i < GBN * GBK / 8; i += 256) {
      int row = i >> 3, cc = (i & 7) << 3;
      *(uint4*)&sB[cur][row][cc] = *(const uint4*)(Bt + (size_t)(bn + row) * K + kt + cc);
    }
    __syncthreads();
    for (int kk = 0; kk < GBK; kk += 32) {
      v16bf a0 = load_fragA(&sA[cur][0][0], GBK, wm, kk);
      v16bf a1 = load_fragA(&sA[cur][0][0], GBK, wm + 16, kk);
      v16bf b0 = load_fragB(&sB[cur][0][0], GBK, wn, kk);
      v16bf b1 = load_fragB(&sB[cur][0][0], GBK, wn + 16, kk);
      acc[0][0] = WMMA_BF16(a0, b0, acc[0][0]);
      acc[0][1] = WMMA_BF16(a0, b1, acc[0][1]);
      acc[1][0] = WMMA_BF16(a1, b0, acc[1][0]);
      acc[1][1] = WMMA_BF16(a1, b1, acc[1][1]);
    }
    __syncthreads();
  }
#endif

  const int lane = tid & 31;
  const int hi8 = (lane >> 4) << 3, cl = lane & 15;
  for (int mt = 0; mt < 2; ++mt)
    for (int nt = 0; nt < 2; ++nt) {
      int col = bn + wn + nt * 16 + cl;
      float bv = bias ? bias[col] : 0.f;
      for (int v = 0; v < 8; ++v) {
        int row = bm + wm + mt * 16 + hi8 + v;
        float xv = acc[mt][nt][v] + bv;
        if (act == 1) {  // tanh-approx GELU (jax default)
          float x3 = xv * xv * xv;
          xv = 0.5f * xv * (1.f + tanhf(0.7978845608028654f * (xv + 0.044715f * x3)));
        }
        if (C)  C[(size_t)row * N + col] = xv;
        if (Cb) Cb[(size_t)row * N + col] = tobf(xv);
      }
    }
}

// ---------------- weight convert + transpose: fp32 [K][N] -> bf16 [N][K] ----------------
__global__ void wconv_kernel(const float* __restrict__ Wsrc, bf16* __restrict__ Wt, int K, int N) {
  size_t idx = (size_t)blockIdx.x * 256 + threadIdx.x;
  if (idx >= (size_t)K * N) return;
  int k = (int)(idx / N), n = (int)(idx % N);
  Wt[(size_t)n * K + k] = tobf(Wsrc[idx]);
}

// ---------------- embedding + LayerNorm (fp32 x + bf16 copy) ----------------
__global__ __launch_bounds__(256) void embed_ln_kernel(
    const int* __restrict__ toks, const float* __restrict__ wemb,
    const float* __restrict__ pemb, const float* __restrict__ temb,
    const float* __restrict__ lw, const float* __restrict__ lb,
    float* __restrict__ x, bf16* __restrict__ xb)
{
  __shared__ float red[256];
  const int row = blockIdx.x;            // b*S + s
  const int s = row % kS;
  const int tid = threadIdx.x;
  const int tok = toks[row];
  float v0[3]; float sum = 0.f;
  for (int i = 0; i < 3; ++i) {
    int d = tid + i * 256;
    float val = wemb[(size_t)tok * kD + d] + pemb[(size_t)s * kD + d] + temb[d];
    v0[i] = val; sum += val;
  }
  red[tid] = sum; __syncthreads();
  for (int off = 128; off > 0; off >>= 1) { if (tid < off) red[tid] += red[tid + off]; __syncthreads(); }
  float mean = red[0] / kD; __syncthreads();
  float sq = 0.f;
  for (int i = 0; i < 3; ++i) { float dd = v0[i] - mean; sq += dd * dd; }
  red[tid] = sq; __syncthreads();
  for (int off = 128; off > 0; off >>= 1) { if (tid < off) red[tid] += red[tid + off]; __syncthreads(); }
  float inv = rsqrtf(red[0] / kD + 1e-5f);
  for (int i = 0; i < 3; ++i) {
    int d = tid + i * 256;
    float o = (v0[i] - mean) * inv * lw[d] + lb[d];
    x[(size_t)row * kD + d] = o;
    xb[(size_t)row * kD + d] = tobf(o);
  }
}

// ---------------- residual add + LayerNorm (in-place x, bf16 copy) ----------------
__global__ __launch_bounds__(256) void add_ln_kernel(
    float* __restrict__ x, const float* __restrict__ a,
    const float* __restrict__ lw, const float* __restrict__ lb,
    bf16* __restrict__ xb)
{
  __shared__ float red[256];
  const size_t row = blockIdx.x;
  const int tid = threadIdx.x;
  float v0[3]; float sum = 0.f;
  for (int i = 0; i < 3; ++i) {
    int d = tid + i * 256;
    float val = x[row * kD + d] + a[row * kD + d];
    v0[i] = val; sum += val;
  }
  red[tid] = sum; __syncthreads();
  for (int off = 128; off > 0; off >>= 1) { if (tid < off) red[tid] += red[tid + off]; __syncthreads(); }
  float mean = red[0] / kD; __syncthreads();
  float sq = 0.f;
  for (int i = 0; i < 3; ++i) { float dd = v0[i] - mean; sq += dd * dd; }
  red[tid] = sq; __syncthreads();
  for (int off = 128; off > 0; off >>= 1) { if (tid < off) red[tid] += red[tid + off]; __syncthreads(); }
  float inv = rsqrtf(red[0] / kD + 1e-5f);
  for (int i = 0; i < 3; ++i) {
    int d = tid + i * 256;
    float o = (v0[i] - mean) * inv * lw[d] + lb[d];
    x[row * kD + d] = o;
    xb[row * kD + d] = tobf(o);
  }
}

// ---------------- pack Q/K/V bf16 into attention layouts ----------------
// Qbf [B,H,S,DH]; Kbf [B,H,SP,DH] zero-padded W rows each side; Vt [B,H,DH,SP] transposed.
__global__ void qkv_pack_kernel(
    const bf16* __restrict__ q, const bf16* __restrict__ k, const bf16* __restrict__ v,
    bf16* __restrict__ Qbf, bf16* __restrict__ Kbf, bf16* __restrict__ Vt)
{
  size_t idx = (size_t)blockIdx.x * blockDim.x + threadIdx.x;
  const size_t total = (size_t)kB * kH * kSP * kDH;
  if (idx >= total) return;
  int d  = idx % kDH;
  int sp = (idx / kDH) % kSP;
  int h  = (idx / ((size_t)kDH * kSP)) % kH;
  int b  = (int)(idx / ((size_t)kDH * kSP * kH));
  bf16 zero = tobf(0.f);
  bf16 kv = zero, vv = zero;
  if (sp >= kW && sp < kW + kS) {
    int s = sp - kW;
    size_t src = ((size_t)(b * kS + s)) * kD + h * kDH + d;
    kv = k[src]; vv = v[src];
    Qbf[(((size_t)(b * kH + h)) * kS + s) * kDH + d] = q[src];
  }
  Kbf[(((size_t)(b * kH + h)) * kSP + sp) * kDH + d] = kv;
  Vt [(((size_t)(b * kH + h)) * kDH + d) * kSP + sp] = vv;
}

// ---------------- local attention (flash-lite per 32-query tile) ----------------
constexpr int kTQ = 32;  // query rows per block; kW/kTQ = 8 blocks per chunk

__global__ __launch_bounds__(256) void attn_kernel(
    const bf16* __restrict__ Qbf, const bf16* __restrict__ Kbf,
    const bf16* __restrict__ Vt, const int* __restrict__ amask,
    bf16* __restrict__ out)
{
  __shared__ __align__(16) bf16  sQ[kTQ][kDH];    // 4 KB
  __shared__ __align__(16) float sS[kTQ][kKW];    // 96 KB
  __shared__ __align__(16) bf16  sP[kTQ][kKW];    // 48 KB
  __shared__ unsigned char sM[kKW];

  const int bx = blockIdx.x;
  const int qt = bx & 7;
  const int c  = (bx >> 3) & (kNCH - 1);
  const int bh = bx >> 7;                // b*H + h
  const int h  = bh % kH;
  const int b  = bh / kH;
  const int tid = threadIdx.x;
  const int lane = tid & 31, wid = tid >> 5;

  const bf16* Qb = Qbf + (((size_t)bh) * kS + c * kW + qt * kTQ) * kDH;
  for (int i = tid; i < kTQ * kDH / 8; i += 256) {
    int row = i / (kDH / 8), cc = (i % (kDH / 8)) * 8;
    *(uint4*)&sQ[row][cc] = *(const uint4*)(Qb + (size_t)row * kDH + cc);
  }
  for (int j = tid; j < kKW; j += 256) {
    int g = c * kW + j;
    int ok = (g >= kW) && (g < kW + kS);
    if (ok) ok = amask[b * kS + (g - kW)] != 0;
    sM[j] = (unsigned char)ok;
  }
  __syncthreads();

  // ---- pass 1: scores = Q @ K^T (masked, scaled) ----
  {
    const bf16* Kb = Kbf + (((size_t)bh) * kSP + c * kW) * kDH;
    v8f acc[2][6] = {};
    const int wn = wid * 96;
    for (int ks = 0; ks < kDH; ks += 32) {
      v16bf a0 = load_fragA(&sQ[0][0], kDH, 0, ks);
      v16bf a1 = load_fragA(&sQ[0][0], kDH, 16, ks);
      for (int nt = 0; nt < 6; ++nt) {
        v16bf bfv = load_fragB(Kb, kDH, wn + nt * 16, ks);
        acc[0][nt] = WMMA_BF16(a0, bfv, acc[0][nt]);
        acc[1][nt] = WMMA_BF16(a1, bfv, acc[1][nt]);
      }
    }
    const int hi8 = (lane >> 4) << 3, cl = lane & 15;
    for (int mt = 0; mt < 2; ++mt)
      for (int nt = 0; nt < 6; ++nt) {
        int j = wn + nt * 16 + cl;
        for (int v = 0; v < 8; ++v) {
          int row = mt * 16 + hi8 + v;
          int qic = qt * kTQ + row;
          int rel = j - qic - kW;
          bool band = (rel >= -kW) && (rel <= kW);
          sS[row][j] = (band && sM[j]) ? acc[mt][nt][v] * 0.125f : -1e9f;
        }
      }
  }
  __syncthreads();

  // ---- softmax: 8 lanes per row ----
  {
    int row = tid >> 3, sl = tid & 7;
    float mx = -3.4e38f;
    for (int j = sl; j < kKW; j += 8) mx = fmaxf(mx, sS[row][j]);
    mx = fmaxf(mx, __shfl_xor(mx, 1, 8));
    mx = fmaxf(mx, __shfl_xor(mx, 2, 8));
    mx = fmaxf(mx, __shfl_xor(mx, 4, 8));
    float sum = 0.f;
    for (int j = sl; j < kKW; j += 8) {
      float e = __expf(sS[row][j] - mx);
      sS[row][j] = e;
      sum += e;
    }
    sum += __shfl_xor(sum, 1, 8);
    sum += __shfl_xor(sum, 2, 8);
    sum += __shfl_xor(sum, 4, 8);
    float inv = 1.f / sum;
    for (int j = sl; j < kKW; j += 8) sP[row][j] = tobf(sS[row][j] * inv);
  }
  __syncthreads();

  // ---- pass 2: out = P @ V  (8 waves = 2 mtiles x 4 ntiles) ----
  {
    const bf16* Vb = Vt + ((size_t)bh) * kDH * kSP + c * kW;
    const int mt = wid >> 2, nt = wid & 3;
    v8f acc = {};
    for (int ks = 0; ks < kKW; ks += 32) {
      v16bf a   = load_fragA(&sP[0][0], kKW, mt * 16, ks);
      v16bf bfv = load_fragB(Vb, kSP, nt * 16, ks);
      acc = WMMA_BF16(a, bfv, acc);
    }
    const int hi8 = (lane >> 4) << 3, cl = lane & 15;
    const int dcol = h * kDH + nt * 16 + cl;
    for (int v = 0; v < 8; ++v) {
      int row = mt * 16 + hi8 + v;
      int s = c * kW + qt * kTQ + row;
      out[((size_t)(b * kS + s)) * kD + dcol] = tobf(acc[v]);
    }
  }
}

// ---------------- mean pool over S ----------------
__global__ void pool_kernel(const float* __restrict__ x, float* __restrict__ pooled) {
  int idx = blockIdx.x * blockDim.x + threadIdx.x;
  if (idx >= kB * kD) return;
  int b = idx / kD, d = idx % kD;
  float s = 0.f;
  for (int t = 0; t < kS; ++t) s += x[((size_t)(b * kS + t)) * kD + d];
  pooled[idx] = s / kS;
}

// ---------------- head: SELU(pooled@p1+b1) @ p2 + b2 ----------------
__global__ __launch_bounds__(512) void head_kernel(
    const float* __restrict__ pooled,
    const float* __restrict__ p1w, const float* __restrict__ p1b,
    const float* __restrict__ p2w, const float* __restrict__ p2b,
    float* __restrict__ out)
{
  __shared__ float sh[512];
  const int b = blockIdx.x, tid = threadIdx.x;
  float acc = p1b[tid];
  for (int d = 0; d < kD; ++d) acc += pooled[b * kD + d] * p1w[(size_t)d * 512 + tid];
  const float sc = 1.0507009873554805f, al = 1.6732632423543772f;
  sh[tid] = sc * (acc > 0.f ? acc : al * (__expf(acc) - 1.f));
  __syncthreads();
  if (tid < 256) {
    float o = p2b[tid];
    for (int j = 0; j < 512; ++j) o += sh[j] * p2w[(size_t)j * 256 + tid];
    out[b * 256 + tid] = o;
  }
}

// ---------------- launcher ----------------
extern "C" void kernel_launch(void* const* d_in, const int* in_sizes, int n_in,
                              void* d_out, int out_size, void* d_ws, size_t ws_size,
                              hipStream_t stream) {
  (void)in_sizes; (void)n_in; (void)out_size; (void)ws_size;
  const int*   toks  = (const int*)d_in[0];
  const int*   amask = (const int*)d_in[1];
  const float* wemb  = (const float*)d_in[2];
  const float* pemb  = (const float*)d_in[3];
  const float* temb  = (const float*)d_in[4];
  const float* lnew  = (const float*)d_in[5];
  const float* lneb  = (const float*)d_in[6];
  const float* Wq = (const float*)d_in[7];   const float* bq = (const float*)d_in[8];
  const float* Wk = (const float*)d_in[9];   const float* bk = (const float*)d_in[10];
  const float* Wv = (const float*)d_in[11];  const float* bv = (const float*)d_in[12];
  const float* Wo = (const float*)d_in[13];  const float* bo = (const float*)d_in[14];
  const float* l1w = (const float*)d_in[15]; const float* l1b = (const float*)d_in[16];
  const float* Wi = (const float*)d_in[17];  const float* bi = (const float*)d_in[18];
  const float* Wf = (const float*)d_in[19];  const float* bfp = (const float*)d_in[20];
  const float* l2w = (const float*)d_in[21]; const float* l2b = (const float*)d_in[22];
  const float* p1w = (const float*)d_in[23]; const float* p1b = (const float*)d_in[24];
  const float* p2w = (const float*)d_in[25]; const float* p2b = (const float*)d_in[26];
  float* outp = (float*)d_out;

  char* ws = (char*)d_ws;
  size_t off = 0;
  auto alloc = [&](size_t bytes) -> void* {
    void* p = ws + off;
    off = (off + bytes + 255) & ~(size_t)255;
    return p;
  };
  const size_t nBSD = (size_t)kB * kS * kD;
  const size_t nPad = (size_t)kB * kH * kSP * kDH;
  float* x      = (float*)alloc(sizeof(float) * nBSD);
  float* tmp    = (float*)alloc(sizeof(float) * nBSD);
  float* pooled = (float*)alloc(sizeof(float) * kB * kD);
  bf16* xbf     = (bf16*)alloc(sizeof(bf16) * nBSD);
  bf16* qb      = (bf16*)alloc(sizeof(bf16) * nBSD);
  bf16* kb      = (bf16*)alloc(sizeof(bf16) * nBSD);
  bf16* vb      = (bf16*)alloc(sizeof(bf16) * nBSD);
  bf16* attnb   = (bf16*)alloc(sizeof(bf16) * nBSD);
  bf16* ffhb    = (bf16*)alloc(sizeof(bf16) * (size_t)kB * kS * kFF);
  bf16* Qbf     = (bf16*)alloc(sizeof(bf16) * nBSD);
  bf16* Kbf     = (bf16*)alloc(sizeof(bf16) * nPad);
  bf16* Vt      = (bf16*)alloc(sizeof(bf16) * nPad);
  bf16* wtq     = (bf16*)alloc(sizeof(bf16) * (size_t)kD * kD);
  bf16* wtk     = (bf16*)alloc(sizeof(bf16) * (size_t)kD * kD);
  bf16* wtv     = (bf16*)alloc(sizeof(bf16) * (size_t)kD * kD);
  bf16* wto     = (bf16*)alloc(sizeof(bf16) * (size_t)kD * kD);
  bf16* wti     = (bf16*)alloc(sizeof(bf16) * (size_t)kD * kFF);  // [FF][D]
  bf16* wtf     = (bf16*)alloc(sizeof(bf16) * (size_t)kFF * kD);  // [D][FF]

  const int M = kB * kS;                          // 8192
  const dim3 blk256(256);
  const dim3 gD(kD / GBN, M / GBM);               // N=768 GEMMs
  const dim3 gFF(kFF / GBN, M / GBM);             // N=3072 GEMM
  const int wDD  = (kD * kD + 255) / 256;         // 2304 blocks
  const int wDFF = (kD * kFF + 255) / 256;        // 9216 blocks

  embed_ln_kernel<<<dim3(kB * kS), blk256, 0, stream>>>(toks, wemb, pemb, temb, lnew, lneb, x, xbf);

  const int packBlocks = (int)(((size_t)kB * kH * kSP * kDH + 255) / 256);
  const int attnBlocks = kB * kH * kNCH * (kW / kTQ);  // 3072

  for (int l = 0; l < kL; ++l) {
    const float* bq_ = bq + (size_t)l * kD;
    const float* bk_ = bk + (size_t)l * kD;
    const float* bv_ = bv + (size_t)l * kD;
    const float* bo_ = bo + (size_t)l * kD;
    const float* bi_ = bi + (size_t)l * kFF;
    const float* bf_ = bfp + (size_t)l * kD;

    // convert + transpose this layer's weights to bf16 [N][K]
    wconv_kernel<<<dim3(wDD),  blk256, 0, stream>>>(Wq + (size_t)l * kD * kD,  wtq, kD, kD);
    wconv_kernel<<<dim3(wDD),  blk256, 0, stream>>>(Wk + (size_t)l * kD * kD,  wtk, kD, kD);
    wconv_kernel<<<dim3(wDD),  blk256, 0, stream>>>(Wv + (size_t)l * kD * kD,  wtv, kD, kD);
    wconv_kernel<<<dim3(wDD),  blk256, 0, stream>>>(Wo + (size_t)l * kD * kD,  wto, kD, kD);
    wconv_kernel<<<dim3(wDFF), blk256, 0, stream>>>(Wi + (size_t)l * kD * kFF, wti, kD, kFF);
    wconv_kernel<<<dim3(wDFF), blk256, 0, stream>>>(Wf + (size_t)l * kFF * kD, wtf, kFF, kD);

    gemm_bf16_kernel<<<gD, blk256, 0, stream>>>(xbf, wtq, bq_, nullptr, qb, M, kD, kD, 0);
    gemm_bf16_kernel<<<gD, blk256, 0, stream>>>(xbf, wtk, bk_, nullptr, kb, M, kD, kD, 0);
    gemm_bf16_kernel<<<gD, blk256, 0, stream>>>(xbf, wtv, bv_, nullptr, vb, M, kD, kD, 0);

    qkv_pack_kernel<<<dim3(packBlocks), blk256, 0, stream>>>(qb, kb, vb, Qbf, Kbf, Vt);

    attn_kernel<<<dim3(attnBlocks), blk256, 0, stream>>>(Qbf, Kbf, Vt, amask, attnb);

    gemm_bf16_kernel<<<gD, blk256, 0, stream>>>(attnb, wto, bo_, tmp, nullptr, M, kD, kD, 0);
    add_ln_kernel<<<dim3(kB * kS), blk256, 0, stream>>>(x, tmp, l1w + (size_t)l * kD, l1b + (size_t)l * kD, xbf);

    gemm_bf16_kernel<<<gFF, blk256, 0, stream>>>(xbf, wti, bi_, nullptr, ffhb, M, kFF, kD, 1);
    gemm_bf16_kernel<<<gD, blk256, 0, stream>>>(ffhb, wtf, bf_, tmp, nullptr, M, kD, kFF, 0);
    add_ln_kernel<<<dim3(kB * kS), blk256, 0, stream>>>(x, tmp, l2w + (size_t)l * kD, l2b + (size_t)l * kD, xbf);
  }

  pool_kernel<<<dim3((kB * kD + 255) / 256), blk256, 0, stream>>>(x, pooled);
  head_kernel<<<dim3(kB), dim3(512), 0, stream>>>(pooled, p1w, p1b, p2w, p2b, outp);
}
